// TokenWise_vMF_KL_CIR_Loss_85504208929142
// MI455X (gfx1250) — matrix operand order repaired and checked by reading, not compiled, not run on hardware
//
#include <hip/hip_runtime.h>

// ---------------------------------------------------------------------------
// TokenWise vMF-KL contrastive loss, gfx1250 (wave32, WMMA).
//
// score = -0.5*(Aq*kq + At*kt) + sim * (Aq*0.5*kt + kq*0.5*At)   (logC cancels)
// Heavy part: 4096x4096x256 GEMM via v_wmma_f32_16x16x32_bf16 with bf16
// hi/lo 3-product decomposition; max-over-Tt / weighted-sum-over-Tq epilogue
// fused (67 MB score tensor never hits memory).
//
// Register blocking: each wave owns a 16x64 C strip (4 accumulators) so the
// A operand is reused 4x -> 12 WMMAs per 20 b128 loads per k-chunk
// (~19 FLOP/byte from cache), instead of 3 WMMAs per 8 loads.
// ---------------------------------------------------------------------------

typedef __attribute__((ext_vector_type(16))) __bf16 v16bf;
typedef __attribute__((ext_vector_type(8)))  float  v8f;
typedef __attribute__((ext_vector_type(4)))  unsigned int u32x4;

union BfLoader {
    u32x4 q[2];   // two 16-byte chunks
    v16bf v;      // 16 bf16 values = 32 bytes
};

#define NTOK 4096         // B*Tq == N*Tt
#define DDIM 256
#define BDIM 128
#define NDIM 128

__device__ __forceinline__ unsigned short f32_to_bf16(float x) {
    unsigned int u = __float_as_uint(x);
    u += 0x7fffu + ((u >> 16) & 1u);     // round to nearest even
    return (unsigned short)(u >> 16);
}
__device__ __forceinline__ float bf16_to_f32(unsigned short h) {
    return __uint_as_float(((unsigned int)h) << 16);
}

// ---------------------------------------------------------------------------
// K1: per-token vMF statistics (Bessel series, 50 terms, online logsumexp).
// ---------------------------------------------------------------------------
__global__ void token_stats_kernel(const float* __restrict__ kq,
                                   const float* __restrict__ kt,
                                   float* __restrict__ aq, float* __restrict__ kqc,
                                   float* __restrict__ u1, float* __restrict__ ent_out,
                                   float* __restrict__ at, float* __restrict__ ktc,
                                   float* __restrict__ v1, float* __restrict__ p1,
                                   float* __restrict__ p2) {
    int i = blockIdx.x * blockDim.x + threadIdx.x;
    if (i >= 2 * NTOK) return;
    bool isT = (i >= NTOK);
    int  j   = isT ? (i - NTOK) : i;
    float kap = isT ? kt[j] : kq[j];
    float kc  = fmaxf(kap, 1e-6f);
    float lx  = logf(kc * 0.5f);

    float m127 = -1e30f, s127 = 0.f;
    float m128 = -1e30f, s128 = 0.f;
    float L1   = 0.0f;               // lgamma(k+1), k=0
    float L128 = lgammaf(128.0f);    // lgamma(k+128), k=0
    for (int k = 0; k < 50; ++k) {
        float kk   = (float)k;
        float L129 = L128 + logf(kk + 128.0f);        // lgamma(k+129)
        float t127 = (2.0f * kk + 127.0f) * lx - (L1 + L128);
        float t128 = (2.0f * kk + 128.0f) * lx - (L1 + L129);
        if (t127 > m127) { s127 = s127 * expf(m127 - t127) + 1.0f; m127 = t127; }
        else             { s127 += expf(t127 - m127); }
        if (t128 > m128) { s128 = s128 * expf(m128 - t128) + 1.0f; m128 = t128; }
        else             { s128 += expf(t128 - m128); }
        L1  += logf(kk + 1.0f);
        L128 = L129;
    }
    float lb127 = m127 + logf(s127);
    float lb128 = m128 + logf(s128);

    float A = expf(fminf(fmaxf(lb128 - lb127, -60.0f), 0.0f));
    A = fminf(fmaxf(A, 1e-8f), 1.0f - 1e-6f);

    if (!isT) {
        aq[j]  = A;
        kqc[j] = kc;
        u1[j]  = -0.5f * A * kc;
        float logC = 127.0f * logf(kc) - 128.0f * 1.8378770664093453f - lb127;
        ent_out[j] = -logC - kc * A;
    } else {
        at[j]  = A;
        ktc[j] = kc;
        v1[j]  = -0.5f * A * kc;
        p1[j]  = 0.5f * kc;
        p2[j]  = 0.5f * A;
    }
}

// ---------------------------------------------------------------------------
// K2: token weights = softmax(-entropy/0.5) over the 32 tokens of each b.
// ---------------------------------------------------------------------------
__global__ void token_weights_kernel(const float* __restrict__ ent,
                                     float* __restrict__ w) {
    int b = blockIdx.x;
    int t = threadIdx.x;   // 0..31
    float x = -2.0f * ent[b * 32 + t];
    float m = x;
    #pragma unroll
    for (int off = 16; off >= 1; off >>= 1)
        m = fmaxf(m, __shfl_xor(m, off, 32));
    float e = expf(x - m);
    float s = e;
    #pragma unroll
    for (int off = 16; off >= 1; off >>= 1)
        s += __shfl_xor(s, off, 32);
    w[b * 32 + t] = e / s;
}

// ---------------------------------------------------------------------------
// K3: split fp32 -> bf16 hi/lo pair (error-compensated decomposition).
// ---------------------------------------------------------------------------
__global__ void split_bf16_kernel(const float* __restrict__ q,
                                  const float* __restrict__ t,
                                  unsigned short* __restrict__ qh,
                                  unsigned short* __restrict__ ql,
                                  unsigned short* __restrict__ th,
                                  unsigned short* __restrict__ tl) {
    const int total = NTOK * DDIM;           // per tensor
    int i = blockIdx.x * blockDim.x + threadIdx.x;
    if (i >= 2 * total) return;
    if (i < total) {
        float x = q[i];
        unsigned short h = f32_to_bf16(x);
        qh[i] = h;
        ql[i] = f32_to_bf16(x - bf16_to_f32(h));
    } else {
        int j = i - total;
        float x = t[j];
        unsigned short h = f32_to_bf16(x);
        th[j] = h;
        tl[j] = f32_to_bf16(x - bf16_to_f32(h));
    }
}

// ---------------------------------------------------------------------------
// K4: fused GEMM + score epilogue, register-blocked.
// Grid = (N/4, B); block = 128 threads = 4 waves.
// Block tile: 32 query rows x 128 target cols (= one b, four n's).
// Wave (wm, wn2) owns rows 16*wm..+15, cols 64*wn2..+63 (4 WMMA C tiles).
// K loop: 8 chunks of 32; per chunk: 1 A(hi/lo) load pair feeds 4 B tiles
// -> 12 WMMAs per 20 b128 loads.
// ---------------------------------------------------------------------------
__global__ void __launch_bounds__(128)
gemm_score_kernel(const unsigned short* __restrict__ qh,
                  const unsigned short* __restrict__ ql,
                  const unsigned short* __restrict__ th,
                  const unsigned short* __restrict__ tl,
                  const float* __restrict__ u1, const float* __restrict__ aq,
                  const float* __restrict__ kqc, const float* __restrict__ v1,
                  const float* __restrict__ p1, const float* __restrict__ p2,
                  const float* __restrict__ w,
                  float* __restrict__ score_out) {
    const int nblk = blockIdx.x;   // covers n = 4*nblk .. 4*nblk+3
    const int b    = blockIdx.y;
    const int tid  = threadIdx.x;
    const int wid  = tid >> 5;     // 0..3
    const int lane = tid & 31;
    const int wm   = wid & 1;      // row quadrant (query tokens)
    const int wn2  = wid >> 1;     // 64-col half (target tokens)
    const int l16  = lane & 15;
    const int h    = lane >> 4;

    // A operand: row = query token (M = l16); elements 0-15 -> K=16h..
    const int rowA = b * 32 + wm * 16 + l16;
    const unsigned short* qhp = qh + rowA * DDIM;
    const unsigned short* qlp = ql + rowA * DDIM;

    // B operand: column token base for tile t is colbase + t*16 rows of t-matrix
    const int coltok0 = nblk * 128 + wn2 * 64 + l16;
    const unsigned short* thp = th + coltok0 * DDIM;
    const unsigned short* tlp = tl + coltok0 * DDIM;
    const int tstride = 16 * DDIM;     // +16 target tokens

    v8f acc[4] = {};

    #pragma unroll
    for (int kc = 0; kc < 8; ++kc) {
        const int k0 = kc * 32;
        BfLoader aHi, aLo;
        aHi.q[0] = *(const u32x4*)(qhp + k0 + 16 * h);
        aHi.q[1] = *(const u32x4*)(qhp + k0 + 16 * h + 8);
        aLo.q[0] = *(const u32x4*)(qlp + k0 + 16 * h);
        aLo.q[1] = *(const u32x4*)(qlp + k0 + 16 * h + 8);
        #pragma unroll
        for (int t = 0; t < 4; ++t) {
            const int off = t * tstride + k0 + 16 * h;
            BfLoader bHi, bLo;
            bHi.q[0] = *(const u32x4*)(thp + off);
            bHi.q[1] = *(const u32x4*)(thp + off + 8);
            bLo.q[0] = *(const u32x4*)(tlp + off);
            bLo.q[1] = *(const u32x4*)(tlp + off + 8);
            acc[t] = __builtin_amdgcn_wmma_f32_16x16x32_bf16(
                false, aHi.v, false, bHi.v, (short)0, acc[t], false, false);
            acc[t] = __builtin_amdgcn_wmma_f32_16x16x32_bf16(
                false, aHi.v, false, bLo.v, (short)0, acc[t], false, false);
            acc[t] = __builtin_amdgcn_wmma_f32_16x16x32_bf16(
                false, aLo.v, false, bHi.v, (short)0, acc[t], false, false);
        }
    }

    // ---- fused epilogue --------------------------------------------------
    // C/D layout: VGPR r, lanes 0-15 -> (M=r, N=l16); lanes 16-31 -> (M=r+8).
    // Tiles t=0,1 belong to local n = 2*wn2; tiles 2,3 to local n = 2*wn2+1.
    __shared__ float lmax[4][32];   // [n_local][row-in-tile]

    float v1c[4], p1c[4], p2c[4];
    #pragma unroll
    for (int t = 0; t < 4; ++t) {
        const int ct = coltok0 + t * 16;
        v1c[t] = v1[ct];
        p1c[t] = p1[ct];
        p2c[t] = p2[ct];
    }

    const int rowbase = b * 32 + wm * 16 + h * 8;
    #pragma unroll
    for (int r = 0; r < 8; ++r) {
        const int row = rowbase + r;
        const float uq  = u1[row];
        const float aqr = aq[row];
        const float kqr = kqc[row];
        float s0 = uq + v1c[0] + acc[0][r] * (aqr * p1c[0] + kqr * p2c[0]);
        float s1 = uq + v1c[1] + acc[1][r] * (aqr * p1c[1] + kqr * p2c[1]);
        float s2 = uq + v1c[2] + acc[2][r] * (aqr * p1c[2] + kqr * p2c[2]);
        float s3 = uq + v1c[3] + acc[3][r] * (aqr * p1c[3] + kqr * p2c[3]);
        float sa = fmaxf(s0, s1);     // first n of this wave
        float sb = fmaxf(s2, s3);     // second n of this wave
        #pragma unroll
        for (int off = 8; off >= 1; off >>= 1) {
            sa = fmaxf(sa, __shfl_xor(sa, off, 16));
            sb = fmaxf(sb, __shfl_xor(sb, off, 16));
        }
        if (l16 == 0) {
            const int rloc = wm * 16 + h * 8 + r;
            lmax[2 * wn2 + 0][rloc] = sa;
            lmax[2 * wn2 + 1][rloc] = sb;
        }
    }
    __syncthreads();

    // Each wave finishes one of the block's four n's: weighted sum over rows.
    {
        const int nl = wid;                      // n_local 0..3
        float val = w[b * 32 + lane] * lmax[nl][lane];
        #pragma unroll
        for (int off = 16; off >= 1; off >>= 1)
            val += __shfl_xor(val, off, 32);
        if (lane == 0)
            score_out[b * NDIM + nblk * 4 + nl] = val;
    }
}

// ---------------------------------------------------------------------------
// K5: logits = score/temp, row log-softmax, per-row loss.
// ---------------------------------------------------------------------------
__global__ void softmax_loss_kernel(const float* __restrict__ score,
                                    const float* __restrict__ temp,
                                    float* __restrict__ logits,
                                    float* __restrict__ loss_each) {
    const int b = blockIdx.x;
    const int n = threadIdx.x;   // 0..127
    __shared__ float red[128];
    __shared__ float diag;

    const float tc = fmaxf(temp[0], 1e-6f);
    const float t  = score[b * NDIM + n] / tc;
    logits[b * NDIM + n] = t;
    if (n == b) diag = t;
    red[n] = t;
    __syncthreads();
    for (int s = 64; s >= 1; s >>= 1) {
        if (n < s) red[n] = fmaxf(red[n], red[n + s]);
        __syncthreads();
    }
    const float m = red[0];
    __syncthreads();
    red[n] = expf(t - m);
    __syncthreads();
    for (int s = 64; s >= 1; s >>= 1) {
        if (n < s) red[n] += red[n + s];
        __syncthreads();
    }
    if (n == 0)
        loss_each[b] = -(diag - (m + logf(red[0])));
}

// ---------------------------------------------------------------------------
// K6: deterministic mean of the 128 per-row losses.
// ---------------------------------------------------------------------------
__global__ void mean_loss_kernel(const float* __restrict__ loss_each,
                                 float* __restrict__ out0) {
    const int t = threadIdx.x;
    __shared__ float red[128];
    red[t] = loss_each[t];
    __syncthreads();
    for (int s = 64; s >= 1; s >>= 1) {
        if (t < s) red[t] += red[t + s];
        __syncthreads();
    }
    if (t == 0) out0[0] = red[0] / 128.0f;
}

// ---------------------------------------------------------------------------
extern "C" void kernel_launch(void* const* d_in, const int* in_sizes, int n_in,
                              void* d_out, int out_size, void* d_ws, size_t ws_size,
                              hipStream_t stream) {
    const float* q    = (const float*)d_in[0];   // (128,32,256)
    const float* kq   = (const float*)d_in[1];   // (128,32)
    const float* t    = (const float*)d_in[2];   // (128,32,256)
    const float* kt   = (const float*)d_in[3];   // (128,32)
    const float* temp = (const float*)d_in[4];   // scalar

    float* out = (float*)d_out;
    float* out_loss   = out;                         // [1]
    float* out_logits = out + 1;                     // [128*128]
    float* out_score  = out + 1 + 16384;             // [128*128]
    float* out_ent    = out + 1 + 16384 + 16384;     // [128*32]

    // ---- workspace layout ----
    char* ws = (char*)d_ws;
    const size_t bf16_sz = (size_t)NTOK * DDIM * sizeof(unsigned short); // 2 MB
    unsigned short* qh = (unsigned short*)(ws + 0 * bf16_sz);
    unsigned short* ql = (unsigned short*)(ws + 1 * bf16_sz);
    unsigned short* th = (unsigned short*)(ws + 2 * bf16_sz);
    unsigned short* tl = (unsigned short*)(ws + 3 * bf16_sz);
    char* fbase = ws + 4 * bf16_sz;
    const size_t fsz = (size_t)NTOK * sizeof(float);                     // 16 KB
    float* aq_a  = (float*)(fbase + 0 * fsz);
    float* kqc_a = (float*)(fbase + 1 * fsz);
    float* u1_a  = (float*)(fbase + 2 * fsz);
    float* w_a   = (float*)(fbase + 3 * fsz);
    float* at_a  = (float*)(fbase + 4 * fsz);
    float* ktc_a = (float*)(fbase + 5 * fsz);
    float* v1_a  = (float*)(fbase + 6 * fsz);
    float* p1_a  = (float*)(fbase + 7 * fsz);
    float* p2_a  = (float*)(fbase + 8 * fsz);
    float* loss_each = (float*)(fbase + 9 * fsz);

    // K1: per-token stats (8192 tokens)
    token_stats_kernel<<<(2 * NTOK + 255) / 256, 256, 0, stream>>>(
        kq, kt, aq_a, kqc_a, u1_a, out_ent, at_a, ktc_a, v1_a, p1_a, p2_a);

    // K2: token weights
    token_weights_kernel<<<BDIM, 32, 0, stream>>>(out_ent, w_a);

    // K3: bf16 hi/lo split of q and t
    {
        int total = 2 * NTOK * DDIM;
        split_bf16_kernel<<<(total + 255) / 256, 256, 0, stream>>>(
            q, t, qh, ql, th, tl);
    }

    // K4: fused WMMA GEMM + score/max/weighted-sum epilogue
    {
        dim3 grid(NDIM / 4, BDIM);
        gemm_score_kernel<<<grid, 128, 0, stream>>>(
            qh, ql, th, tl, u1_a, aq_a, kqc_a, v1_a, p1_a, p2_a, w_a, out_score);
    }

    // K5: logits + log-softmax + per-row loss
    softmax_loss_kernel<<<BDIM, NDIM, 0, stream>>>(out_score, temp,
                                                   out_logits, loss_each);

    // K6: mean loss
    mean_loss_kernel<<<1, 128, 0, stream>>>(loss_each, out_loss);
}
